// GraphSAGEConv_65592740544796
// MI455X (gfx1250) — matrix operand order repaired and checked by reading, not compiled, not run on hardware
//
#include <hip/hip_runtime.h>
#include <hip/hip_bf16.h>

typedef __attribute__((ext_vector_type(16))) __bf16 v16bf;
typedef __attribute__((ext_vector_type(8)))  float  v8f;

#define N_NODES 8192
#define FEAT    128
#define BATCH   4
#define MT      128
#define KTILE   32
#define NSTEPS  (N_NODES / KTILE)   // 256

// ---- bf16 helpers (round-to-nearest-even) ----
__device__ __forceinline__ unsigned bfr(float f) {
    unsigned u = __float_as_uint(f);
    return (u + 0x7FFFu + ((u >> 16) & 1u)) >> 16;
}
__device__ __forceinline__ float ubf(unsigned h) { return __uint_as_float(h << 16); }
// pack two adjacency entries into a bf16{0,1} pair (even K in [15:0])
__device__ __forceinline__ unsigned mask2(int a, int b) {
    return (a > 0 ? 0x3F80u : 0u) | (b > 0 ? 0x3F800000u : 0u);
}
// flat shared pointer -> 32-bit LDS byte address (addr[31:0] per aperture rules)
__device__ __forceinline__ unsigned lds_addr32(const void* p) {
    return (unsigned)(unsigned long long)p;
}

// ---------------------------------------------------------------------------
// prep_x: x[b][j][f] f32 -> transposed bf16 hi/lo XT[b][f][j]
// ---------------------------------------------------------------------------
__global__ __launch_bounds__(256) void prep_x_kernel(
    const float* __restrict__ x, unsigned short* __restrict__ xh,
    unsigned short* __restrict__ xl)
{
    const int t = threadIdx.x;
    const int f = t & 127;
    const int jo = t >> 7;            // 0..1
    const int blk = blockIdx.x;       // 0..255
    const int b = blk >> 6;
    const int jbase = (blk & 63) * 128;
    for (int i = 0; i < 64; ++i) {
        const int j = jbase + jo * 64 + i;
        const float v = x[((long)b * N_NODES + j) * FEAT + f];
        const unsigned h = bfr(v);
        const unsigned l = bfr(v - ubf(h));
        const long o = ((long)b * FEAT + f) * N_NODES + j;
        xh[o] = (unsigned short)h;
        xl[o] = (unsigned short)l;
    }
}

// ---------------------------------------------------------------------------
// prep_w: Wcat = [W_self; W_neigh] (256x128) -> bf16 hi/lo, pre-swizzled into
// WMMA B-fragment order: u32 index = ((kt*8 + ft)*32 + lane)*8 + v
// ---------------------------------------------------------------------------
__global__ __launch_bounds__(256) void prep_w_kernel(
    const float* __restrict__ wself, const float* __restrict__ wneigh,
    unsigned* __restrict__ wh, unsigned* __restrict__ wl)
{
    const int tid = blockIdx.x * 256 + threadIdx.x;   // 0..16383
    const int v  = tid & 7;
    const int L  = (tid >> 3) & 31;
    const int ft = (tid >> 8) & 7;
    const int kt = tid >> 11;                          // 0..7
    const int n  = L & 15;
    const int kb = (L < 16) ? 0 : 8;
    const int f  = ft * 16 + n;
    const int k  = kt * 32 + kb + 2 * (v & 3) + ((v & 4) ? 16 : 0);
    const float a0 = (k < 128)     ? wself[k * FEAT + f]       : wneigh[(k - 128) * FEAT + f];
    const float a1 = (k + 1 < 128) ? wself[(k + 1) * FEAT + f] : wneigh[(k + 1 - 128) * FEAT + f];
    const unsigned h0 = bfr(a0), h1 = bfr(a1);
    const unsigned l0 = bfr(a0 - ubf(h0)), l1 = bfr(a1 - ubf(h1));
    wh[tid] = h0 | (h1 << 16);
    wl[tid] = l0 | (l1 << 16);
}

// ---------------------------------------------------------------------------
// deg_k: deg[i] = #(adj[i][j] > 0), one block per row
// ---------------------------------------------------------------------------
__global__ __launch_bounds__(256) void deg_kernel(
    const int* __restrict__ adj, float* __restrict__ deg)
{
    __shared__ int red[256];
    const int row = blockIdx.x;
    const int* rp = adj + (long)row * N_NODES;
    int c = 0;
    for (int j = threadIdx.x; j < N_NODES; j += 256) c += (rp[j] > 0);
    red[threadIdx.x] = c;
    __syncthreads();
    for (int s = 128; s > 0; s >>= 1) {
        if (threadIdx.x < s) red[threadIdx.x] += red[threadIdx.x + s];
        __syncthreads();
    }
    if (threadIdx.x == 0) deg[row] = (float)red[0];
}

// ---------------------------------------------------------------------------
// agg_gemm: agg[b][i][f] = (sum_j mask[i][j] * x[b][j][f]) / max(deg,1)
// WG = 128 rows x 128 feats, 8 waves, double-buffered LDS, bf16 WMMA.
// B (x hi/lo) tiles staged with GLOBAL_LOAD_ASYNC_TO_LDS_B128 (ASYNCcnt);
// A (mask) tiles converted int32->bf16{0,1} through VGPRs (DScnt).
// ---------------------------------------------------------------------------
__global__ __launch_bounds__(256) void agg_gemm_kernel(
    const int* __restrict__ adj, const unsigned short* __restrict__ xthi,
    const unsigned short* __restrict__ xtlo, const float* __restrict__ deg,
    float* __restrict__ agg)
{
    __shared__ __align__(16) unsigned a_sm[2][128 * 16];  // 128 rows x 32 bf16
    __shared__ __align__(16) unsigned bh_sm[2][2048];     // fragment order
    __shared__ __align__(16) unsigned bl_sm[2][2048];

    const int t    = threadIdx.x;
    const int lane = t & 31;
    const int wave = t >> 5;
    const int i0   = blockIdx.x * MT;
    const int b    = blockIdx.y;

    // A staging: thread -> (row, half of 16 ints)
    const int  arow  = t >> 1;
    const int  ahalf = t & 1;
    const long abase = (long)(i0 + arow) * N_NODES + ahalf * 16;

    // B staging: thread -> one (ft, lane) fragment slice = 16 bf16 contiguous j
    const int  ft_s = t >> 5;
    const int  Ls   = t & 31;
    const int  kb_s = (Ls < 16) ? 0 : 16;
    const long xrow = ((long)b * FEAT + (ft_s * 16 + (Ls & 15))) * N_NODES + kb_s;

    v8f acc[8] = {};

    auto stage = [&](int kt, int buf) {
        const int j0 = kt * KTILE;
        // ---- A: load 16 ints, convert to bf16 {0,1}, store 32B to LDS ----
        const int4* ap = (const int4*)(adj + abase + j0);
        int4 q0 = ap[0], q1 = ap[1], q2 = ap[2], q3 = ap[3];
        uint4 p0, p1;
        p0.x = mask2(q0.x, q0.y); p0.y = mask2(q0.z, q0.w);
        p0.z = mask2(q1.x, q1.y); p0.w = mask2(q1.z, q1.w);
        p1.x = mask2(q2.x, q2.y); p1.y = mask2(q2.z, q2.w);
        p1.z = mask2(q3.x, q3.y); p1.w = mask2(q3.z, q3.w);
        uint4* asp = (uint4*)&a_sm[buf][arow * 16 + ahalf * 8];
        asp[0] = p0; asp[1] = p1;

        // ---- B: async DMA global->LDS, 32B per thread per matrix ----
        const unsigned long long gh = (unsigned long long)(xthi + xrow + j0);
        const unsigned long long gl = (unsigned long long)(xtlo + xrow + j0);
        const unsigned dh = lds_addr32(&bh_sm[buf][t * 8]);
        const unsigned dl = lds_addr32(&bl_sm[buf][t * 8]);
        asm volatile(
            "global_load_async_to_lds_b128 %0, %2, off\n\t"
            "global_load_async_to_lds_b128 %0, %2, off offset:16\n\t"
            "global_load_async_to_lds_b128 %1, %3, off\n\t"
            "global_load_async_to_lds_b128 %1, %3, off offset:16"
            :: "v"(dh), "v"(dl), "v"(gh), "v"(gl)
            : "memory");

        if (kt + 1 < NSTEPS)  // gfx1250 global_prefetch_b8 on next adj tile
            __builtin_prefetch((const void*)(adj + abase + j0 + KTILE), 0, 3);
    };

    auto compute = [&](int buf) {
        union { v16bf v; uint4 q[2]; } A;
        const int kb4 = (lane < 16) ? 0 : 8;
        const unsigned* ab = &a_sm[buf][(wave * 16 + (lane & 15)) * 16];
        A.q[0] = *(const uint4*)(ab + kb4);
        A.q[1] = *(const uint4*)(ab + 4 + kb4);
#pragma unroll
        for (int ft = 0; ft < 8; ++ft) {
            union { v16bf v; uint4 q[2]; } Bh, Bl;
            const unsigned* bh = &bh_sm[buf][(ft * 32 + lane) * 8];
            const unsigned* bl = &bl_sm[buf][(ft * 32 + lane) * 8];
            Bh.q[0] = *(const uint4*)(bh); Bh.q[1] = *(const uint4*)(bh + 4);
            Bl.q[0] = *(const uint4*)(bl); Bl.q[1] = *(const uint4*)(bl + 4);
            acc[ft] = __builtin_amdgcn_wmma_f32_16x16x32_bf16(
                false, A.v, false, Bh.v, (short)0, acc[ft], false, false);
            acc[ft] = __builtin_amdgcn_wmma_f32_16x16x32_bf16(
                false, A.v, false, Bl.v, (short)0, acc[ft], false, false);
        }
    };

    stage(0, 0);
    asm volatile("s_wait_asynccnt 0x0" ::: "memory");
    __syncthreads();
    for (int kt = 0; kt < NSTEPS; ++kt) {
        if (kt + 1 < NSTEPS) stage(kt + 1, (kt + 1) & 1);
        compute(kt & 1);
        // each wave drains its own async stores before publishing via barrier
        asm volatile("s_wait_asynccnt 0x0" ::: "memory");
        __syncthreads();
    }

    // epilogue: scale by 1/max(deg,1), store f32
    const int mrow0 = i0 + wave * 16 + ((lane < 16) ? 0 : 8);
    float rd[8];
#pragma unroll
    for (int r = 0; r < 8; ++r) {
        float d = deg[mrow0 + r];
        rd[r] = 1.0f / (d > 1.0f ? d : 1.0f);
    }
#pragma unroll
    for (int ft = 0; ft < 8; ++ft) {
        const int n = ft * 16 + (lane & 15);
        float* orow = agg + ((long)b * N_NODES + mrow0) * FEAT + n;
#pragma unroll
        for (int r = 0; r < 8; ++r) orow[(long)r * FEAT] = acc[ft][r] * rd[r];
    }
}

// ---------------------------------------------------------------------------
// out_gemm: out = relu(x@Ws + bs + [deg>0]*(agg@Wn + bn)) as one K=256 GEMM
// over A = [x | agg] rows (B*N = 32768), 3-term bf16 split for f32 accuracy.
// ---------------------------------------------------------------------------
__global__ __launch_bounds__(256) void out_gemm_kernel(
    const float* __restrict__ x, const float* __restrict__ agg,
    const unsigned* __restrict__ wfh, const unsigned* __restrict__ wfl,
    const float* __restrict__ bs, const float* __restrict__ bn,
    const float* __restrict__ deg, float* __restrict__ out)
{
    const int t = threadIdx.x, lane = t & 31, wave = t >> 5;
    const long g0 = (long)blockIdx.x * 128;
    const long grow = g0 + wave * 16 + (lane & 15);
    const int kbase = (lane < 16) ? 0 : 8;

    v8f acc[8] = {};
    const float* arow_x = x   + grow * FEAT;
    const float* arow_a = agg + grow * FEAT;

#pragma unroll
    for (int kt = 0; kt < 8; ++kt) {
        const float* src = (kt < 4) ? (arow_x + kt * 32) : (arow_a + (kt - 4) * 32);
        float4 f0 = *(const float4*)(src + kbase);
        float4 f1 = *(const float4*)(src + kbase + 4);
        float4 f2 = *(const float4*)(src + kbase + 16);
        float4 f3 = *(const float4*)(src + kbase + 20);
        float va[16] = { f0.x, f0.y, f0.z, f0.w, f1.x, f1.y, f1.z, f1.w,
                         f2.x, f2.y, f2.z, f2.w, f3.x, f3.y, f3.z, f3.w };
        union { v16bf v; unsigned u[8]; } Ah, Al;
#pragma unroll
        for (int i = 0; i < 8; ++i) {
            const float a0 = va[2 * i], a1 = va[2 * i + 1];
            const unsigned h0 = bfr(a0), h1 = bfr(a1);
            const unsigned l0 = bfr(a0 - ubf(h0)), l1 = bfr(a1 - ubf(h1));
            Ah.u[i] = h0 | (h1 << 16);
            Al.u[i] = l0 | (l1 << 16);
        }
#pragma unroll
        for (int ft = 0; ft < 8; ++ft) {
            union { v16bf v; uint4 q[2]; } Bh, Bl;
            const long idx = (((long)kt * 8 + ft) * 32 + lane) * 8;
            Bh.q[0] = *(const uint4*)(wfh + idx); Bh.q[1] = *(const uint4*)(wfh + idx + 4);
            Bl.q[0] = *(const uint4*)(wfl + idx); Bl.q[1] = *(const uint4*)(wfl + idx + 4);
            acc[ft] = __builtin_amdgcn_wmma_f32_16x16x32_bf16(
                false, Ah.v, false, Bh.v, (short)0, acc[ft], false, false);
            acc[ft] = __builtin_amdgcn_wmma_f32_16x16x32_bf16(
                false, Ah.v, false, Bl.v, (short)0, acc[ft], false, false);
            acc[ft] = __builtin_amdgcn_wmma_f32_16x16x32_bf16(
                false, Al.v, false, Bh.v, (short)0, acc[ft], false, false);
        }
    }

    // epilogue: bias + deg mask + relu
    const long m0 = g0 + wave * 16 + ((lane < 16) ? 0 : 8);
    float degpos[8];
#pragma unroll
    for (int r = 0; r < 8; ++r)
        degpos[r] = (deg[(int)((m0 + r) & (N_NODES - 1))] > 0.0f) ? 1.0f : 0.0f;
#pragma unroll
    for (int ft = 0; ft < 8; ++ft) {
        const int n = ft * 16 + (lane & 15);
        const float bsv = bs[n], bnv = bn[n];
#pragma unroll
        for (int r = 0; r < 8; ++r) {
            float v = acc[ft][r] + bsv + degpos[r] * bnv;
            out[(m0 + r) * FEAT + n] = v > 0.0f ? v : 0.0f;
        }
    }
}

// ---------------------------------------------------------------------------
extern "C" void kernel_launch(void* const* d_in, const int* in_sizes, int n_in,
                              void* d_out, int out_size, void* d_ws, size_t ws_size,
                              hipStream_t stream) {
    const float* x      = (const float*)d_in[0];
    const int*   adj    = (const int*)d_in[1];
    const float* wself  = (const float*)d_in[2];
    const float* bself  = (const float*)d_in[3];
    const float* wneigh = (const float*)d_in[4];
    const float* bneigh = (const float*)d_in[5];
    float* out = (float*)d_out;

    char* ws = (char*)d_ws;
    const size_t XT_BYTES = (size_t)BATCH * FEAT * N_NODES * 2;  // 8 MB each
    unsigned short* xh  = (unsigned short*)(ws);
    unsigned short* xl  = (unsigned short*)(ws + XT_BYTES);
    unsigned*       wh  = (unsigned*)(ws + 2 * XT_BYTES);
    unsigned*       wl  = (unsigned*)(ws + 2 * XT_BYTES + 65536);
    float*          deg = (float*)(ws + 2 * XT_BYTES + 2 * 65536);
    float*          agg = (float*)(ws + 2 * XT_BYTES + 2 * 65536 + 32768);

    prep_x_kernel<<<256, 256, 0, stream>>>(x, xh, xl);
    prep_w_kernel<<<64, 256, 0, stream>>>(wself, wneigh, wh, wl);
    deg_kernel<<<N_NODES, 256, 0, stream>>>(adj, deg);
    agg_gemm_kernel<<<dim3(64, 4), 256, 0, stream>>>(adj, xh, xl, deg, agg);
    out_gemm_kernel<<<256, 256, 0, stream>>>(x, agg, wh, wl, bself, bneigh, deg, out);
}
